// LSTM_SiLU_86466281603876
// MI455X (gfx1250) — compile-verified, matrix operand order checked by reading, not run on hardware
//
#include <hip/hip_runtime.h>

// ---------------------------------------------------------------------------
// LSTM + SiLU gate, CDNA5 (gfx1250), wave32, bf16 WMMA with f32 accumulation.
//   T=1024, B=32, I=1024, H=1024
// Phase 1: convert/transpose weights to bf16; precompute xW = x@W + b and
//          xG = x@Wg_x + bg with two big WMMA GEMMs (parallel over T*B rows).
// Phase 2: ONE persistent kernel runs all 1024 recurrent steps. 32 blocks x
//          128 threads stay resident; a device-scope atomic grid barrier
//          (1 per step) replaces 2048 kernel launches. R^T (8MB bf16) and
//          Wg_h^T (2MB) stay hot in the 192MB L2 across all steps.
// ---------------------------------------------------------------------------

#define T_DIM 1024
#define B_DIM 32
#define I_DIM 1024
#define H_DIM 1024
#define REC_BLOCKS 32   // persistent grid: 32 blocks x 128 threads = 128 waves

typedef unsigned short u16;
typedef __bf16 bf16;
typedef bf16  v16bf __attribute__((ext_vector_type(16)));
typedef float v8f   __attribute__((ext_vector_type(8)));

// round-to-nearest-even f32 -> bf16 (bit pattern)
__device__ __forceinline__ u16 f2bf(float f) {
  unsigned int u = __float_as_uint(f);
  unsigned int r = u + 0x7FFFu + ((u >> 16) & 1u);
  return (u16)(r >> 16);
}

__device__ __forceinline__ float sigmoidf_(float x) {
  return 1.0f / (1.0f + __expf(-x));
}

// Load a 16x32 bf16 WMMA fragment (A layout; B uses the same code on
// transposed weights). rowk = row base + k0 (k0 multiple of 32). Per ISA
// layout: lane = {m = lane&15, kh = lane>>4}; lane holds K in
// [8*kh, 8*kh+8) and [16+8*kh, 16+8*kh+8): two contiguous 16B chunks.
__device__ __forceinline__ v16bf load_frag(const u16* rowk, int kh) {
  union { uint4 q[2]; v16bf v; } u;
  u.q[0] = *(const uint4*)(rowk + 8 * kh);
  u.q[1] = *(const uint4*)(rowk + 16 + 8 * kh);
  return u.v;
}

__device__ __forceinline__ v8f wmma_bf16(v16bf a, v16bf b, v8f c) {
  // (neg_a, A, neg_b, B, c_mod, C, reuse_a, reuse_b)
  return __builtin_amdgcn_wmma_f32_16x16x32_bf16(false, a, false, b,
                                                 (short)0, c, false, false);
}

// Device-scope grid barrier (classic count+generation). Safe: grid is 32
// co-resident blocks. Counters re-initialized by init kernel every call.
__device__ __forceinline__ void grid_barrier(int* count, int* gen, int nblk) {
  __syncthreads();
  if (threadIdx.x == 0) {
    __threadfence();  // make this block's global writes visible device-wide
    int g = __hip_atomic_load(gen, __ATOMIC_ACQUIRE, __HIP_MEMORY_SCOPE_AGENT);
    if (atomicAdd(count, 1) == nblk - 1) {
      __hip_atomic_store(count, 0, __ATOMIC_RELAXED, __HIP_MEMORY_SCOPE_AGENT);
      __hip_atomic_store(gen, g + 1, __ATOMIC_RELEASE, __HIP_MEMORY_SCOPE_AGENT);
    } else {
      while (__hip_atomic_load(gen, __ATOMIC_ACQUIRE,
                               __HIP_MEMORY_SCOPE_AGENT) == g) {
        __builtin_amdgcn_s_sleep(1);
      }
    }
    __threadfence();
  }
  __syncthreads();
}

// ------------------------------ prep kernels --------------------------------

__global__ void init_barrier(int* p) {
  if (threadIdx.x < 2) p[threadIdx.x] = 0;
}

__global__ void convert_f32_bf16(const float* __restrict__ src,
                                 u16* __restrict__ dst, size_t n) {
  size_t i = (size_t)blockIdx.x * blockDim.x + threadIdx.x;
  if (i < n) dst[i] = f2bf(src[i]);
}

__global__ void copy_f32(const float* __restrict__ src,
                         float* __restrict__ dst, size_t n) {
  size_t i = (size_t)blockIdx.x * blockDim.x + threadIdx.x;
  if (i < n) dst[i] = src[i];
}

// src is [K, N] f32 (row-major), dst is [N, K] bf16 (row-major).
__global__ void transpose_bf16(const float* __restrict__ src,
                               u16* __restrict__ dst, int K, int N) {
  size_t i = (size_t)blockIdx.x * blockDim.x + threadIdx.x;
  size_t total = (size_t)K * N;
  if (i < total) {
    int k = (int)(i / N);
    int n = (int)(i % N);
    dst[(size_t)n * K + k] = f2bf(src[i]);
  }
}

// ----------------------------- big input GEMM -------------------------------
// C[M,N] = A_bf16[M,K] @ Bt_bf16[N,K]^T + bias[N], f32 out.
// Block = 256 threads = 8 waves covering a 32(M) x 64(N) tile.
// Grid = (N/64, M/32).
__global__ __launch_bounds__(256) void gemm_bf16_bias(
    const u16* __restrict__ A, const u16* __restrict__ Bt,
    const float* __restrict__ bias, float* __restrict__ C,
    int M, int N, int K) {
  const int lane = threadIdx.x & 31;
  const int wave = threadIdx.x >> 5;
  const int kh = lane >> 4;
  const int l15 = lane & 15;
  const int mbase = blockIdx.y * 32 + (wave & 1) * 16;
  const int nbase = blockIdx.x * 64 + (wave >> 1) * 16;

  const u16* arow = A + (size_t)(mbase + l15) * K;
  const u16* brow = Bt + (size_t)(nbase + l15) * K;

  v8f acc = {};
  for (int k = 0; k < K; k += 32) {
    v16bf a = load_frag(arow + k, kh);
    v16bf b = load_frag(brow + k, kh);
    acc = wmma_bf16(a, b, acc);
  }
  // C/D layout: lane -> column (lane&15), VGPR r -> row 8*(lane>>4)+r.
  const int col = nbase + l15;
  const float bv = bias[col];
#pragma unroll
  for (int r = 0; r < 8; ++r) {
    int row = mbase + 8 * kh + r;
    C[(size_t)row * N + col] = acc[r] + bv;
  }
}

// ------------------------- persistent recurrence ----------------------------
// 32 blocks x 128 threads (4 waves: wave&1 = M tile, wave>>1 = 16-col group).
// Each block owns 32 columns of H. Per timestep:
//   Phase A: v = xW_t + h@R -> gates -> c_new, h_new (4 WMMA chains/wave)
//   [grid barrier]  (B needs ALL columns of h_new)
//   Phase B: gate = silu(xG_t + h_new@Wg_h); y_t = h_new * gate
// Ping-ponging h_bf AND h_f32 makes B(t) / A(t+1) hazard-free -> exactly one
// barrier per step.
__global__ __launch_bounds__(128) void lstm_recurrence(
    const u16* __restrict__ Rt,      // [4H,H] bf16 = R^T
    const u16* __restrict__ WghT,    // [H,H]  bf16 = Wg_h^T
    const float* __restrict__ xW,    // [T,B,4H] f32 (bias folded)
    const float* __restrict__ xG,    // [T,B,H]  f32 (bias folded)
    u16* __restrict__ hbf0, u16* __restrict__ hbf1,   // [B,H] bf16 ping-pong
    float* __restrict__ hf0, float* __restrict__ hf1, // [B,H] f32 ping-pong
    float* __restrict__ cbuf,        // [B,H] f32 (thread-owned, in-place)
    float* __restrict__ y_out,       // [T,B,H]
    float* __restrict__ hn_out,      // [B,H]
    float* __restrict__ cn_out,      // [B,H]
    int* __restrict__ bar) {         // [2] count, gen
  const int H = H_DIM, B = B_DIM, T = T_DIM;
  const int lane = threadIdx.x & 31;
  const int wave = threadIdx.x >> 5;
  const int kh = lane >> 4;
  const int l15 = lane & 15;
  const int mbase = (wave & 1) * 16;                       // B=32 -> 2 M tiles
  const int col = blockIdx.x * 32 + (wave >> 1) * 16 + l15;

  // Weight row pointers are invariant across timesteps.
  const u16* rowi = Rt + (size_t)(0 * H + col) * H;
  const u16* rowg = Rt + (size_t)(1 * H + col) * H;
  const u16* rowf = Rt + (size_t)(2 * H + col) * H;
  const u16* rowo = Rt + (size_t)(3 * H + col) * H;
  const u16* rowB = WghT + (size_t)col * H;

  u16* hbf[2] = {hbf0, hbf1};
  float* hf[2] = {hf0, hf1};

  for (int t = 0; t < T; ++t) {
    const u16* h_in = hbf[t & 1];
    u16* h_out = hbf[(t + 1) & 1];
    float* hf_out = hf[(t + 1) & 1];
    const float* xW_t = xW + (size_t)t * B * 4 * H;

    // ---- Phase A: gates ----
    const u16* arow = h_in + (size_t)(mbase + l15) * H;
    v8f ai = {}, ag = {}, af = {}, ao = {};
    for (int k = 0; k < H; k += 32) {
      v16bf a = load_frag(arow + k, kh);
      ai = wmma_bf16(a, load_frag(rowi + k, kh), ai);
      ag = wmma_bf16(a, load_frag(rowg + k, kh), ag);
      af = wmma_bf16(a, load_frag(rowf + k, kh), af);
      ao = wmma_bf16(a, load_frag(rowo + k, kh), ao);
    }
#pragma unroll
    for (int r = 0; r < 8; ++r) {
      int row = mbase + 8 * kh + r;  // batch index
      size_t o4 = (size_t)row * (4 * H);
      float vi = ai[r] + xW_t[o4 + 0 * H + col];
      float vg = ag[r] + xW_t[o4 + 1 * H + col];
      float vf = af[r] + xW_t[o4 + 2 * H + col];
      float vo = ao[r] + xW_t[o4 + 3 * H + col];
      float ig = sigmoidf_(vi);
      float gg = tanhf(vg);
      float fg = sigmoidf_(vf);
      float og = sigmoidf_(vo);
      size_t idx = (size_t)row * H + col;
      float cn = fg * cbuf[idx] + ig * gg;
      float hn = og * tanhf(cn);
      cbuf[idx] = cn;
      hf_out[idx] = hn;
      h_out[idx] = f2bf(hn);
      if (t == T - 1) {
        hn_out[idx] = hn;
        cn_out[idx] = cn;
      }
    }

    grid_barrier(bar, bar + 1, REC_BLOCKS);

    // ---- Phase B: SiLU selectivity gate ----
    const u16* arow2 = h_out + (size_t)(mbase + l15) * H;
    v8f acc = {};
    for (int k = 0; k < H; k += 32) {
      v16bf a = load_frag(arow2 + k, kh);
      v16bf b = load_frag(rowB + k, kh);
      acc = wmma_bf16(a, b, acc);
    }
    const float* xG_t = xG + (size_t)t * B * H;
    float* y_t = y_out + (size_t)t * B * H;
#pragma unroll
    for (int r = 0; r < 8; ++r) {
      int row = mbase + 8 * kh + r;
      size_t idx = (size_t)row * H + col;
      float u = acc[r] + xG_t[idx];
      float gate = u * sigmoidf_(u);  // SiLU
      y_t[idx] = hf_out[idx] * gate;
    }
    // No barrier here: A(t+1) touches only the other ping-pong buffers and
    // thread-owned c; all reads it conflicts with were fenced by sync(t).
  }
}

// ------------------------------- launcher -----------------------------------

extern "C" void kernel_launch(void* const* d_in, const int* in_sizes, int n_in,
                              void* d_out, int out_size, void* d_ws,
                              size_t ws_size, hipStream_t stream) {
  (void)in_sizes; (void)n_in; (void)out_size; (void)ws_size;
  const int T = T_DIM, B = B_DIM, I = I_DIM, H = H_DIM;

  const float* x   = (const float*)d_in[0];  // [T,B,I]
  const float* h0  = (const float*)d_in[1];  // [B,H]
  const float* c0  = (const float*)d_in[2];  // [B,H]
  const float* W   = (const float*)d_in[3];  // [I,4H]
  const float* R   = (const float*)d_in[4];  // [H,4H]
  const float* bb  = (const float*)d_in[5];  // [4H]
  const float* Wgx = (const float*)d_in[6];  // [I,H]
  const float* Wgh = (const float*)d_in[7];  // [H,H]
  const float* bg  = (const float*)d_in[8];  // [H]

  float* y_out  = (float*)d_out;                    // [T,B,H]
  float* hn_out = y_out + (size_t)T * B * H;        // [B,H]
  float* cn_out = hn_out + (size_t)B * H;           // [B,H]

  // Workspace carve-up (256B aligned).
  char* ws = (char*)d_ws;
  size_t off = 0;
  auto carve = [&](size_t bytes) {
    char* p = ws + off;
    off += (bytes + 255) & ~(size_t)255;
    return p;
  };
  u16*   x_bf  = (u16*)carve((size_t)T * B * I * 2);      //  64 MB
  u16*   Wt    = (u16*)carve((size_t)4 * H * I * 2);      //   8 MB
  u16*   Rt    = (u16*)carve((size_t)4 * H * H * 2);      //   8 MB
  u16*   WgxT  = (u16*)carve((size_t)H * I * 2);          //   2 MB
  u16*   WghT  = (u16*)carve((size_t)H * H * 2);          //   2 MB
  float* xW    = (float*)carve((size_t)T * B * 4 * H * 4);// 512 MB
  float* xG    = (float*)carve((size_t)T * B * H * 4);    // 128 MB
  u16*   hbf0  = (u16*)carve((size_t)B * H * 2);
  u16*   hbf1  = (u16*)carve((size_t)B * H * 2);
  float* hf0   = (float*)carve((size_t)B * H * 4);
  float* hf1   = (float*)carve((size_t)B * H * 4);
  float* cbuf  = (float*)carve((size_t)B * H * 4);
  int*   bar   = (int*)carve(256);                        // barrier count+gen

  // ---- Phase 1: conversions / transposes / barrier init ----
  init_barrier<<<dim3(1), 64, 0, stream>>>(bar);
  {
    size_t n = (size_t)T * B * I;
    convert_f32_bf16<<<dim3((unsigned)((n + 255) / 256)), 256, 0, stream>>>(x, x_bf, n);
  }
  {
    size_t n = (size_t)I * 4 * H;
    transpose_bf16<<<dim3((unsigned)((n + 255) / 256)), 256, 0, stream>>>(W, Wt, I, 4 * H);
  }
  {
    size_t n = (size_t)H * 4 * H;
    transpose_bf16<<<dim3((unsigned)((n + 255) / 256)), 256, 0, stream>>>(R, Rt, H, 4 * H);
  }
  {
    size_t n = (size_t)I * H;
    transpose_bf16<<<dim3((unsigned)((n + 255) / 256)), 256, 0, stream>>>(Wgx, WgxT, I, H);
  }
  {
    size_t n = (size_t)H * H;
    transpose_bf16<<<dim3((unsigned)((n + 255) / 256)), 256, 0, stream>>>(Wgh, WghT, H, H);
  }
  {
    size_t n = (size_t)B * H;
    convert_f32_bf16<<<dim3((unsigned)((n + 255) / 256)), 256, 0, stream>>>(h0, hbf0, n);
    copy_f32<<<dim3((unsigned)((n + 255) / 256)), 256, 0, stream>>>(c0, cbuf, n);
  }

  // ---- Phase 1b: input projections (fully parallel over T*B) ----
  {
    int M = T * B;  // 32768
    gemm_bf16_bias<<<dim3(4 * H / 64, M / 32), 256, 0, stream>>>(
        x_bf, Wt, bb, xW, M, 4 * H, I);
    gemm_bf16_bias<<<dim3(H / 64, M / 32), 256, 0, stream>>>(
        x_bf, WgxT, bg, xG, M, H, I);
  }

  // ---- Phase 2: single persistent kernel for the whole recurrence ----
  lstm_recurrence<<<dim3(REC_BLOCKS), 128, 0, stream>>>(
      Rt, WghT, xW, xG, hbf0, hbf1, hf0, hf1, cbuf,
      y_out, hn_out, cn_out, bar);
}
